// BpsMlp_33715493273844
// MI455X (gfx1250) — matrix operand order repaired and checked by reading, not compiled, not run on hardware
//
#include <hip/hip_runtime.h>
#include <hip/hip_bf16.h>
#include <cstddef>

typedef _Float16 f16;
typedef __attribute__((ext_vector_type(16))) _Float16 v16h;
typedef __attribute__((ext_vector_type(2)))  _Float16 v2h;
typedef __attribute__((ext_vector_type(8)))  float    v8f;

#define BB   64
#define NN   1024
#define MM   4096
#define HIDD 2048
#define OUTD 512

// ---------------------------------------------------------------------------
// basis4[m] = (bx, by, bz, ||b||^2)
// ---------------------------------------------------------------------------
__global__ void prep_basis_kernel(const float* __restrict__ basis,
                                  float4* __restrict__ basis4) {
    int m = blockIdx.x * blockDim.x + threadIdx.x;
    if (m >= MM) return;
    float bx = basis[m * 3 + 0];
    float by = basis[m * 3 + 1];
    float bz = basis[m * 3 + 2];
    basis4[m] = make_float4(bx, by, bz, bx * bx + by * by + bz * bz);
}

// ---------------------------------------------------------------------------
// x[b,n] = sqrt(max(min_m d2, 1e-12))  (sqrt/min commute: min of monotone fn)
// emitted as f16 row-major -> GEMM0's A matrix
// ---------------------------------------------------------------------------
__global__ void min_dist_kernel(const float* __restrict__ pos,
                                const float4* __restrict__ basis4,
                                f16* __restrict__ x16) {
    int t = blockIdx.x * blockDim.x + threadIdx.x;   // t = b*NN + n
    if (t >= BB * NN) return;
    float px = pos[t * 3 + 0];
    float py = pos[t * 3 + 1];
    float pz = pos[t * 3 + 2];
    float p2 = px * px + py * py + pz * pz;
    float dmin = 3.4e38f;
#pragma unroll 4
    for (int m = 0; m < MM; ++m) {
        float4 q = basis4[m];
        float s  = fmaf(px, q.x, fmaf(py, q.y, pz * q.z));
        float d2 = fmaf(-2.0f, s, p2 + q.w);
        dmin = fminf(dmin, d2);
    }
    x16[t] = (f16)sqrtf(fmaxf(dmin, 1e-12f));
}

// ---------------------------------------------------------------------------
// GEMM: C[64 x Ndim] = act(A[64 x Kdim] @ W[Kdim x Ndim] + bias)
//
// Block = 4 waves; blockIdx.x = 16-column output strip. Each wave owns Kdim/4
// of the K reduction and all four 16-row M-tiles. W stays in its natural
// row-major f32 form and is gathered/converted to the WMMA B-fragment layout
// in registers: for fragment element e, lanes 0-15 read row (k0+e) at 16
// consecutive columns (64 B contiguous) and lanes 16-31 read row (k0+16+e) --
// every weight element is read exactly once per GEMM (46 MB total, the floor).
// Partial accumulators are reduced across waves through LDS; wave 0 applies
// bias (+ReLU) and writes the next layer's f16 activation (or f32 final out).
// ---------------------------------------------------------------------------
template <int Kdim, int Ndim, bool RELU, bool LAST>
__global__ void wmma_gemm_kernel(const f16* __restrict__ A,
                                 const float* __restrict__ W,
                                 const float* __restrict__ bias,
                                 f16* __restrict__ Cf16,
                                 float* __restrict__ Cf32) {
    __shared__ float red[4][4][8][32];          // [wave][mtile][row][lane]

    const int nt   = blockIdx.x;
    const int w    = threadIdx.x >> 5;
    const int lane = threadIdx.x & 31;
    const int l15  = lane & 15;
    const int hi   = lane >> 4;
    const int n    = nt * 16 + l15;

    const v8f zero = {0.f, 0.f, 0.f, 0.f, 0.f, 0.f, 0.f, 0.f};
    v8f acc[4];
#pragma unroll
    for (int mt = 0; mt < 4; ++mt) acc[mt] = zero;

    constexpr int KC = Kdim / 4;                // K chunk per wave

    // per-lane column pointer into W for this wave's K chunk
    const float* wptr = W + (size_t)(w * KC + 16 * hi) * Ndim + n;

    for (int k0 = w * KC; k0 < (w + 1) * KC; k0 += 32) {
        // B fragment: gather 16 K-slices of this column, f32 -> f16 in regs
        v16h bfrag;
#pragma unroll
        for (int e = 0; e < 16; ++e)
            bfrag[e] = (f16)wptr[(size_t)e * Ndim];

        // stream ahead: prefetch next K-block of W
        if (k0 + 32 < (w + 1) * KC)
            __builtin_prefetch(wptr + (size_t)32 * Ndim, 0, 1);
        wptr += (size_t)32 * Ndim;

#pragma unroll
        for (int mt = 0; mt < 4; ++mt) {
            // A fragment: 16x32 f16, contiguous k = 16*hi + e per lane, same as B
            const f16* arow = A + (size_t)(mt * 16 + l15) * Kdim + k0;
            v16h afrag;
#pragma unroll
            for (int j = 0; j < 8; ++j) {
                const int kp = 2 * j + 16 * hi;
                v2h h2 = *(const v2h*)(arow + kp);
                afrag[2 * j]     = h2[0];
                afrag[2 * j + 1] = h2[1];
            }
            acc[mt] = __builtin_amdgcn_wmma_f32_16x16x32_f16(
                false, afrag, false, bfrag, (short)0, acc[mt], false, false);
        }
    }

    // cross-wave K reduction through LDS
#pragma unroll
    for (int mt = 0; mt < 4; ++mt)
#pragma unroll
        for (int r = 0; r < 8; ++r)
            red[w][mt][r][lane] = acc[mt][r];
    __syncthreads();

    if (w == 0) {
        const float bn = bias[n];
#pragma unroll
        for (int mt = 0; mt < 4; ++mt) {
#pragma unroll
            for (int r = 0; r < 8; ++r) {
                float s = red[0][mt][r][lane] + red[1][mt][r][lane]
                        + red[2][mt][r][lane] + red[3][mt][r][lane] + bn;
                if (RELU) s = fmaxf(s, 0.f);
                const int m = mt * 16 + r + 8 * hi;   // C/D layout: lanes 16-31 -> M+8
                if (LAST)
                    Cf32[(size_t)m * Ndim + n] = s;
                else
                    Cf16[(size_t)m * Ndim + n] = (f16)s;
            }
        }
    }
}

// ---------------------------------------------------------------------------
extern "C" void kernel_launch(void* const* d_in, const int* in_sizes, int n_in,
                              void* d_out, int out_size, void* d_ws, size_t ws_size,
                              hipStream_t stream) {
    const float* pos   = (const float*)d_in[0];
    const float* basis = (const float*)d_in[1];
    const float* W0    = (const float*)d_in[2];
    const float* b0    = (const float*)d_in[3];
    const float* W1    = (const float*)d_in[4];
    const float* b1    = (const float*)d_in[5];
    const float* W2    = (const float*)d_in[6];
    const float* b2    = (const float*)d_in[7];
    const float* W3    = (const float*)d_in[8];
    const float* b3    = (const float*)d_in[9];
    float* out = (float*)d_out;

    char*  ws  = (char*)d_ws;
    size_t off = 0;
    auto carve = [&](size_t bytes) -> char* {
        off = (off + 255) & ~(size_t)255;
        char* p = ws + off;
        off += bytes;
        return p;
    };
    float4* basis4 = (float4*)carve((size_t)MM * sizeof(float4));
    f16* x16 = (f16*)carve((size_t)BB * NN   * sizeof(f16));
    f16* a1  = (f16*)carve((size_t)BB * HIDD * sizeof(f16));
    f16* a2  = (f16*)carve((size_t)BB * HIDD * sizeof(f16));
    f16* a3  = (f16*)carve((size_t)BB * HIDD * sizeof(f16));
    (void)ws_size; (void)in_sizes; (void)n_in; (void)out_size;

    prep_basis_kernel<<<(MM + 255) / 256, 256, 0, stream>>>(basis, basis4);
    min_dist_kernel<<<(BB * NN + 255) / 256, 256, 0, stream>>>(pos, basis4, x16);

    wmma_gemm_kernel<NN,   HIDD, true,  false><<<HIDD / 16, 128, 0, stream>>>(x16, W0, b0, a1, nullptr);
    wmma_gemm_kernel<HIDD, HIDD, true,  false><<<HIDD / 16, 128, 0, stream>>>(a1,  W1, b1, a2, nullptr);
    wmma_gemm_kernel<HIDD, HIDD, true,  false><<<HIDD / 16, 128, 0, stream>>>(a2,  W2, b2, a3, nullptr);
    wmma_gemm_kernel<HIDD, OUTD, false, true ><<<OUTD / 16, 128, 0, stream>>>(a3,  W3, b3, nullptr, out);
}
